// GQA_FlashAttn_42219528519935
// MI455X (gfx1250) — compile-verified, matrix operand order checked by reading, not run on hardware
//
#include <hip/hip_runtime.h>
#include <hip/hip_bf16.h>
#include <stdint.h>

typedef __bf16 bf16_t;
typedef __attribute__((ext_vector_type(16))) __bf16 v16bf;
typedef __attribute__((ext_vector_type(8)))  __bf16 v8bf;
typedef __attribute__((ext_vector_type(4)))  __bf16 v4bf;
typedef __attribute__((ext_vector_type(8)))  float  v8f;
typedef __attribute__((ext_vector_type(4)))  unsigned int u32x4;
typedef __attribute__((ext_vector_type(4)))  int   i32x4;
typedef __attribute__((ext_vector_type(8)))  int   i32x8;

#define SEQ_LEN 2048
#define DIM_IN  2048
#define NH      16
#define NKVH    4
#define HD      128

// Build a 16-element bf16 WMMA fragment from two contiguous 8-element (16B) runs.
// Matches CDNA5 16-bit A/B layout: lane-half selects K subgroup (hw*8.., 16+hw*8..).
__device__ __forceinline__ v16bf frag16(const bf16_t* p0, const bf16_t* p1) {
  union { v16bf v; v8bf h[2]; } u;
  u.h[0] = *(const v8bf*)p0;
  u.h[1] = *(const v8bf*)p1;
  return u.v;
}

__device__ __forceinline__ v8f wmma_bf16(v16bf a, v16bf b, v8f c) {
  return __builtin_amdgcn_wmma_f32_16x16x32_bf16(false, a, false, b, (short)0, c,
                                                 false, false);
}

// ---------------------------------------------------------------------------
// TDM: DMA one [tile_rows=128] x [tile_k=32] bf16 tile (row stride = K elems)
// from global memory into LDS at lds_addr.  Descriptor per CDNA5 D# spec.
// ---------------------------------------------------------------------------
__device__ __forceinline__ void tdm_load_tile(uint32_t lds_addr, const bf16_t* g,
                                              uint32_t rows_total, uint32_t Kdim) {
  uint64_t ga = (uint64_t)(uintptr_t)g;
  u32x4 g0;
  g0[0] = 1u;                                   // count=1, user descriptor
  g0[1] = lds_addr;                             // LDS byte address
  g0[2] = (uint32_t)ga;                         // global_addr[31:0]
  g0[3] = ((uint32_t)(ga >> 32) & 0x01FFFFFFu)  // global_addr[56:32]
        | (2u << 30);                           // type = 2 ("image")
  i32x8 g1;
  g1[0] = (int)0x00010000u;                     // wg_mask=0, data_size=1 (2B)
  g1[1] = (int)((Kdim & 0xFFFFu) << 16);        // tensor_dim0[15:0]
  g1[2] = (int)(((Kdim >> 16) & 0xFFFFu) |      // tensor_dim0[31:16]
                ((rows_total & 0xFFFFu) << 16)); // tensor_dim1[15:0]
  g1[3] = (int)(((rows_total >> 16) & 0xFFFFu) | // tensor_dim1[31:16]
                (32u << 16));                   // tile_dim0 = 32
  g1[4] = (int)128u;                            // tile_dim1 = 128, tile_dim2 = 0
  g1[5] = (int)Kdim;                            // tensor_dim0_stride[31:0]
  g1[6] = 0;                                    // stride[47:32], dim1_stride lo
  g1[7] = 0;
  i32x4 gz = {0, 0, 0, 0};
#if defined(__clang_major__) && __clang_major__ >= 23
  i32x8 gz8 = {0, 0, 0, 0, 0, 0, 0, 0};
  __builtin_amdgcn_tensor_load_to_lds(g0, g1, gz, gz, gz8, 0);
#else
  __builtin_amdgcn_tensor_load_to_lds(g0, g1, gz, gz, 0);
#endif
}

// ---------------------------------------------------------------------------
// fp32 -> bf16 converter (vectorized)
// ---------------------------------------------------------------------------
__global__ void cvt_f32_bf16_k(const float* __restrict__ in,
                               bf16_t* __restrict__ out, int n4) {
  int i = blockIdx.x * blockDim.x + threadIdx.x;
  if (i >= n4) return;
  float4 f = ((const float4*)in)[i];
  v4bf o;
  o[0] = (bf16_t)f.x; o[1] = (bf16_t)f.y; o[2] = (bf16_t)f.z; o[3] = (bf16_t)f.w;
  ((v4bf*)out)[i] = o;
}

// ---------------------------------------------------------------------------
// Transpose + convert: in[K][N] f32 -> out[N][K] bf16 (LDS-tiled, 32x32).
// Both GEMM operands become K-major so TDM can stage them identically.
// ---------------------------------------------------------------------------
__launch_bounds__(256)
__global__ void transpose_cvt_k(const float* __restrict__ in,
                                bf16_t* __restrict__ out, int K, int N) {
  __shared__ bf16_t tile[32][33];
  const int tx = threadIdx.x & 31;
  const int ty = threadIdx.x >> 5;      // 0..7
  const int n0 = blockIdx.x * 32;
  const int k0 = blockIdx.y * 32;
#pragma unroll
  for (int i = 0; i < 32; i += 8)
    tile[ty + i][tx] = (bf16_t)in[(size_t)(k0 + ty + i) * N + n0 + tx];
  __syncthreads();
#pragma unroll
  for (int i = 0; i < 32; i += 8)
    out[(size_t)(n0 + ty + i) * K + k0 + tx] = tile[tx][ty + i];
}

// ---------------------------------------------------------------------------
// Tiled bf16 WMMA GEMM with TDM double-buffered staging.
// C[M][N] f32 = A[M][K] @ Bt[N][K]^T   (both operands K-major bf16)
// Block 256 = 8 waves; tile 128x128x32; wave grid 2(M) x 4(N).
// ---------------------------------------------------------------------------
__launch_bounds__(256)
__global__ void gemm_bf16_k(const bf16_t* __restrict__ A,
                            const bf16_t* __restrict__ Bt,
                            float* __restrict__ C, int M, int N, int K) {
  __shared__ __align__(128) bf16_t Ash[2][128][32];
  __shared__ __align__(128) bf16_t Bsh[2][128][32];

  const int tid  = threadIdx.x;
  const int lane = tid & 31;
  const int wid  = tid >> 5;
  const int r    = lane & 15;
  const int hw   = lane >> 4;
  const int wm   = wid >> 2;      // 0..1
  const int wn   = wid & 3;       // 0..3
  const int mBase = blockIdx.y * 128;
  const int nBase = blockIdx.x * 128;

  const uint32_t ldsA0 = (uint32_t)(uintptr_t)&Ash[0][0][0];
  const uint32_t ldsA1 = (uint32_t)(uintptr_t)&Ash[1][0][0];
  const uint32_t ldsB0 = (uint32_t)(uintptr_t)&Bsh[0][0][0];
  const uint32_t ldsB1 = (uint32_t)(uintptr_t)&Bsh[1][0][0];

  const v8f zero = {0.f, 0.f, 0.f, 0.f, 0.f, 0.f, 0.f, 0.f};
  v8f acc[4][2];
#pragma unroll
  for (int mi = 0; mi < 4; ++mi)
#pragma unroll
    for (int ni = 0; ni < 2; ++ni) acc[mi][ni] = zero;

  const int nk = K >> 5;

  // Prologue: DMA tile 0 into buffer 0.
  if (wid == 0) {
    tdm_load_tile(ldsA0, A  + (size_t)mBase * K, (uint32_t)M, (uint32_t)K);
    tdm_load_tile(ldsB0, Bt + (size_t)nBase * K, (uint32_t)N, (uint32_t)K);
    __builtin_amdgcn_s_wait_tensorcnt(0);
  }
  __syncthreads();

  for (int kt = 0; kt < nk; ++kt) {
    const int cur = kt & 1;
    // Kick off DMA for the next tile into the other buffer (overlaps WMMAs).
    if ((kt + 1 < nk) && wid == 0) {
      const int k0n = (kt + 1) << 5;
      tdm_load_tile(cur ? ldsA0 : ldsA1, A  + (size_t)mBase * K + k0n,
                    (uint32_t)M, (uint32_t)K);
      tdm_load_tile(cur ? ldsB0 : ldsB1, Bt + (size_t)nBase * K + k0n,
                    (uint32_t)N, (uint32_t)K);
    }

    v16bf af[4], bfr[2];
#pragma unroll
    for (int mi = 0; mi < 4; ++mi) {
      int row = wm * 64 + mi * 16 + r;
      af[mi] = frag16(&Ash[cur][row][hw * 8], &Ash[cur][row][16 + hw * 8]);
    }
#pragma unroll
    for (int ni = 0; ni < 2; ++ni) {
      int col = wn * 32 + ni * 16 + r;
      bfr[ni] = frag16(&Bsh[cur][col][hw * 8], &Bsh[cur][col][16 + hw * 8]);
    }
#pragma unroll
    for (int mi = 0; mi < 4; ++mi)
#pragma unroll
      for (int ni = 0; ni < 2; ++ni)
        acc[mi][ni] = wmma_bf16(af[mi], bfr[ni], acc[mi][ni]);

    if (wid == 0) __builtin_amdgcn_s_wait_tensorcnt(0);
    __syncthreads();
  }

  // C/D layout: lane r = col, vgpr v -> row v + 8*hw.
#pragma unroll
  for (int mi = 0; mi < 4; ++mi)
#pragma unroll
    for (int ni = 0; ni < 2; ++ni)
#pragma unroll
      for (int v = 0; v < 8; ++v) {
        int mrow = mBase + wm * 64 + mi * 16 + v + 8 * hw;
        int ncol = nBase + wn * 32 + ni * 16 + r;
        C[(size_t)mrow * N + ncol] = acc[mi][ni][v];
      }
}

// ---------------------------------------------------------------------------
// Per-head LayerNorm + RoPE.  One wave per (seq, head) row of 128 values.
// In:  X[s][H*128] fp32.  Out: bf16 [h][s][128] (head-major for attention).
// ---------------------------------------------------------------------------
__launch_bounds__(128)
__global__ void ln_rope_k(const float* __restrict__ X,
                          const float* __restrict__ cosT,
                          const float* __restrict__ sinT,
                          const float* __restrict__ gamma,
                          const float* __restrict__ beta,
                          bf16_t* __restrict__ out, int H) {
  const int wrow = blockIdx.x * 4 + (threadIdx.x >> 5);
  const int lane = threadIdx.x & 31;
  const int s = wrow / H;
  const int h = wrow % H;
  const float* row = X + (size_t)s * (H * HD) + h * HD;

  float4 t = *(const float4*)(row + lane * 4);
  float x[4] = {t.x, t.y, t.z, t.w};

  float sum = x[0] + x[1] + x[2] + x[3];
#pragma unroll
  for (int m = 1; m < 32; m <<= 1) sum += __shfl_xor(sum, m, 32);
  const float mu = sum * (1.0f / 128.0f);

  float vs = 0.f;
#pragma unroll
  for (int i = 0; i < 4; ++i) { float d = x[i] - mu; vs += d * d; }
#pragma unroll
  for (int m = 1; m < 32; m <<= 1) vs += __shfl_xor(vs, m, 32);
  const float rinv = rsqrtf(vs * (1.0f / 128.0f) + 1e-6f);

  float n[4];
#pragma unroll
  for (int i = 0; i < 4; ++i) {
    int d = lane * 4 + i;
    n[i] = (x[i] - mu) * rinv * gamma[d] + beta[d];
  }
  // RoPE partner element (d ^ 64) lives in lane ^ 16, same sub-index.
  const float sgn = (lane < 16) ? -1.0f : 1.0f;
  v4bf o;
#pragma unroll
  for (int i = 0; i < 4; ++i) {
    int d = lane * 4 + i;
    float p = __shfl_xor(n[i], 16, 32);
    float r2 = n[i] * cosT[s * HD + d] + sgn * p * sinT[s * HD + d];
    o[i] = (bf16_t)r2;
  }
  *(v4bf*)&out[((size_t)h * SEQ_LEN + s) * HD + lane * 4] = o;
}

// ---------------------------------------------------------------------------
// V: fp32 [s][KVH*128] -> bf16 transposed [kvh][d][s].
// ---------------------------------------------------------------------------
__global__ void vtrans_k(const float* __restrict__ Vf, bf16_t* __restrict__ Vt) {
  int i = blockIdx.x * blockDim.x + threadIdx.x;   // KVH*128*SEQ total
  int s = i & (SEQ_LEN - 1);
  int rest = i >> 11;
  int d = rest & (HD - 1);
  int kvh = rest >> 7;
  Vt[((size_t)kvh * HD + d) * SEQ_LEN + s] =
      (bf16_t)Vf[(size_t)s * (NKVH * HD) + kvh * HD + d];
}

// ---------------------------------------------------------------------------
// Causal flash attention. One wave per (16-query tile, head).
// Q:[h][s][128] bf16, K:[kvh][s][128] bf16, Vt:[kvh][d][s] bf16.
// ---------------------------------------------------------------------------
__launch_bounds__(32)
__global__ void attn_k(const bf16_t* __restrict__ Qb,
                       const bf16_t* __restrict__ Kb,
                       const bf16_t* __restrict__ Vt,
                       bf16_t* __restrict__ AO) {
  __shared__ __align__(16) bf16_t Plds[16][32];

  const int qt  = blockIdx.x;
  const int h   = blockIdx.y;
  const int kvh = h >> 2;               // GQA group = 4
  const int lane = threadIdx.x;
  const int r  = lane & 15;
  const int hw = lane >> 4;
  const int qbase = qt * 16;

  const bf16_t* Qrow = Qb + ((size_t)h * SEQ_LEN + qbase + r) * HD;
  v16bf aq[4];
#pragma unroll
  for (int kc = 0; kc < 4; ++kc)
    aq[kc] = frag16(Qrow + kc * 32 + hw * 8, Qrow + kc * 32 + 16 + hw * 8);

  const v8f zero = {0.f, 0.f, 0.f, 0.f, 0.f, 0.f, 0.f, 0.f};
  v8f O[8];
  float mrow[8], lrow[8];
#pragma unroll
  for (int ni = 0; ni < 8; ++ni) O[ni] = zero;
#pragma unroll
  for (int v = 0; v < 8; ++v) { mrow[v] = -1e30f; lrow[v] = 0.f; }

  const float scale = 0.08838834764831845f;  // 1/sqrt(128)
  const int nsteps = (qbase + 16 + 31) >> 5; // 32 keys per step

  for (int jt = 0; jt < nsteps; ++jt) {
    const int jb = jt * 32;
    const bf16_t* Kr0 = Kb + ((size_t)kvh * SEQ_LEN + jb + r) * HD;
    const bf16_t* Kr1 = Kr0 + 16 * HD;
    if (jt + 1 < nsteps)  // pull next K tile toward L0/L2
      __builtin_prefetch(Kr0 + 32 * HD, 0, 1);

    v8f s0 = zero, s1 = zero;
#pragma unroll
    for (int kc = 0; kc < 4; ++kc) {
      v16bf bk0 = frag16(Kr0 + kc * 32 + hw * 8, Kr0 + kc * 32 + 16 + hw * 8);
      s0 = wmma_bf16(aq[kc], bk0, s0);
      v16bf bk1 = frag16(Kr1 + kc * 32 + hw * 8, Kr1 + kc * 32 + 16 + hw * 8);
      s1 = wmma_bf16(aq[kc], bk1, s1);
    }
    float p0[8], p1[8], alpha[8];
#pragma unroll
    for (int v = 0; v < 8; ++v) {
      const int q = qbase + v + 8 * hw;
      float a = s0[v] * scale; if (jb + r      > q) a = -1e30f;
      float b = s1[v] * scale; if (jb + 16 + r > q) b = -1e30f;
      float mx = fmaxf(a, b);
#pragma unroll
      for (int msk = 1; msk < 16; msk <<= 1) mx = fmaxf(mx, __shfl_xor(mx, msk, 32));
      const float mnew = fmaxf(mrow[v], mx);
      const float al = __expf(mrow[v] - mnew);
      const float e0 = __expf(a - mnew);
      const float e1 = __expf(b - mnew);
      float rs = e0 + e1;
#pragma unroll
      for (int msk = 1; msk < 16; msk <<= 1) rs += __shfl_xor(rs, msk, 32);
      lrow[v] = lrow[v] * al + rs;
      mrow[v] = mnew;
      alpha[v] = al;
      p0[v] = e0; p1[v] = e1;
    }
    // C-layout -> A-layout for P via LDS bounce.
#pragma unroll
    for (int v = 0; v < 8; ++v) {
      Plds[v + 8 * hw][r]      = (bf16_t)p0[v];
      Plds[v + 8 * hw][16 + r] = (bf16_t)p1[v];
    }
    __syncthreads();
    v16bf pa = frag16(&Plds[r][hw * 8], &Plds[r][16 + hw * 8]);

    const bf16_t* Vbase = Vt + (size_t)kvh * HD * SEQ_LEN;
#pragma unroll
    for (int ni = 0; ni < 8; ++ni) {
      const bf16_t* vp = Vbase + (size_t)(ni * 16 + r) * SEQ_LEN + jb;
      v16bf bv = frag16(vp + hw * 8, vp + 16 + hw * 8);
      v8f cs;
#pragma unroll
      for (int v = 0; v < 8; ++v) cs[v] = O[ni][v] * alpha[v];
      O[ni] = wmma_bf16(pa, bv, cs);
    }
    __syncthreads();
  }

#pragma unroll
  for (int ni = 0; ni < 8; ++ni)
#pragma unroll
    for (int v = 0; v < 8; ++v) {
      const int q = qbase + v + 8 * hw;
      AO[(size_t)q * (NH * HD) + h * HD + ni * 16 + r] =
          (bf16_t)(O[ni][v] / lrow[v]);
    }
}

// ---------------------------------------------------------------------------
extern "C" void kernel_launch(void* const* d_in, const int* in_sizes, int n_in,
                              void* d_out, int out_size, void* d_ws, size_t ws_size,
                              hipStream_t stream) {
  (void)in_sizes; (void)n_in; (void)out_size; (void)ws_size;
  const float* x     = (const float*)d_in[0];
  const float* cosT  = (const float*)d_in[1];
  const float* sinT  = (const float*)d_in[2];
  const float* Wq    = (const float*)d_in[3];
  const float* Wk    = (const float*)d_in[4];
  const float* Wv    = (const float*)d_in[5];
  const float* Wo    = (const float*)d_in[6];
  const float* qgam  = (const float*)d_in[7];
  const float* qbeta = (const float*)d_in[8];
  const float* kgam  = (const float*)d_in[9];
  const float* kbeta = (const float*)d_in[10];
  float* out = (float*)d_out;

  char* ws = (char*)d_ws;
  const size_t MB = 1ull << 20;
  bf16_t* xb   = (bf16_t*)(ws +  0 * MB);  //  8 MB  x          [2048][2048]
  bf16_t* Wqt  = (bf16_t*)(ws +  8 * MB);  //  8 MB  Wq^T       [2048][2048]
  bf16_t* Wkt  = (bf16_t*)(ws + 16 * MB);  //  2 MB  Wk^T       [512][2048]
  bf16_t* Wvt  = (bf16_t*)(ws + 18 * MB);  //  2 MB  Wv^T       [512][2048]
  bf16_t* Wot  = (bf16_t*)(ws + 20 * MB);  //  8 MB  Wo^T       [2048][2048]
  float*  Qf   = (float*) (ws + 28 * MB);  // 16 MB
  float*  Kf   = (float*) (ws + 44 * MB);  //  4 MB
  float*  Vf   = (float*) (ws + 48 * MB);  //  4 MB
  bf16_t* Qr   = (bf16_t*)(ws + 52 * MB);  //  8 MB
  bf16_t* Kr   = (bf16_t*)(ws + 60 * MB);  //  2 MB
  bf16_t* Vt   = (bf16_t*)(ws + 62 * MB);  //  2 MB
  bf16_t* AOb  = (bf16_t*)(ws + 64 * MB);  //  8 MB  (72 MB total)

  // 1) operand conversion; weights transposed to K-major for TDM staging
  cvt_f32_bf16_k <<<4096, 256, 0, stream>>>(x, xb, 1048576);
  transpose_cvt_k<<<dim3(64, 64), 256, 0, stream>>>(Wq, Wqt, DIM_IN, NH * HD);
  transpose_cvt_k<<<dim3(16, 64), 256, 0, stream>>>(Wk, Wkt, DIM_IN, NKVH * HD);
  transpose_cvt_k<<<dim3(16, 64), 256, 0, stream>>>(Wv, Wvt, DIM_IN, NKVH * HD);
  transpose_cvt_k<<<dim3(64, 64), 256, 0, stream>>>(Wo, Wot, DIM_IN, DIM_IN);

  // 2) QKV projections (WMMA + TDM staging)
  gemm_bf16_k<<<dim3(16, 16), 256, 0, stream>>>(xb, Wqt, Qf, SEQ_LEN, NH * HD,   DIM_IN);
  gemm_bf16_k<<<dim3( 4, 16), 256, 0, stream>>>(xb, Wkt, Kf, SEQ_LEN, NKVH * HD, DIM_IN);
  gemm_bf16_k<<<dim3( 4, 16), 256, 0, stream>>>(xb, Wvt, Vf, SEQ_LEN, NKVH * HD, DIM_IN);

  // 3) LN + RoPE, V transpose
  ln_rope_k<<<(SEQ_LEN * NH)   / 4, 128, 0, stream>>>(Qf, cosT, sinT, qgam, qbeta, Qr, NH);
  ln_rope_k<<<(SEQ_LEN * NKVH) / 4, 128, 0, stream>>>(Kf, cosT, sinT, kgam, kbeta, Kr, NKVH);
  vtrans_k<<<(NKVH * HD * SEQ_LEN) / 256, 256, 0, stream>>>(Vf, Vt);

  // 4) causal flash attention (WMMA)
  attn_k<<<dim3(SEQ_LEN / 16, NH), 32, 0, stream>>>(Qr, Kr, Vt, AOb);

  // 5) output projection (WMMA + TDM) -> fp32 d_out
  gemm_bf16_k<<<dim3(16, 16), 256, 0, stream>>>(AOb, Wot, out, SEQ_LEN, DIM_IN, DIM_IN);
}
